// CrossAttention_8040178778622
// MI455X (gfx1250) — matrix operand order repaired and checked by reading, not compiled, not run on hardware
//
#include <hip/hip_runtime.h>

typedef __attribute__((ext_vector_type(16))) _Float16 v16h;
typedef __attribute__((ext_vector_type(8)))  _Float16 v8h;
typedef __attribute__((ext_vector_type(8)))  float    v8f;

#define BQ 4
#define FQ 1024
#define TQ 1024
#define CQ 1024
#define HQ 16
#define DQ 64

static __device__ __forceinline__ v16h make_frag(v8h lo, v8h hi) {
  v16h r;
#pragma unroll
  for (int e = 0; e < 8; ++e) { r[e] = lo[e]; r[e + 8] = hi[e]; }
  return r;
}

static __device__ __forceinline__ v8f wmma_f16(v16h a, v16h b, v8f c) {
  // D = A(16x32 f16) x B(32x16 f16) + C(16x16 f32)
  return __builtin_amdgcn_wmma_f32_16x16x32_f16(false, a, false, b,
                                                (short)0, c, false, false);
}

// gfx1250 async global->LDS copy, 16 bytes per lane (tracked by ASYNCcnt)
static __device__ __forceinline__ void async_copy16(_Float16* lds_dst,
                                                    const _Float16* gsrc) {
  unsigned ldsa = (unsigned)(unsigned long long)lds_dst;  // low 32b = LDS offset
  asm volatile("global_load_async_to_lds_b128 %0, %1, off"
               :: "v"(ldsa), "v"(gsrc) : "memory");
}
static __device__ __forceinline__ void wait_async0() {
  asm volatile("s_wait_asynccnt 0" ::: "memory");
}

// ---------------------------------------------------------------------------
// Kernel 0: one-shot f32 -> f16 conversion (inputs and weights), so all GEMM
// tile staging becomes pure async global->LDS copies.
// ---------------------------------------------------------------------------
__global__ __launch_bounds__(256) void cvt_f16_kernel(
    const float* __restrict__ src, _Float16* __restrict__ dst, int n4) {
  int stride = gridDim.x * blockDim.x;
  for (int i = blockIdx.x * blockDim.x + threadIdx.x; i < n4; i += stride) {
    float4 t = *(const float4*)(src + (size_t)i * 4);
    _Float16* p = dst + (size_t)i * 4;
    p[0] = (_Float16)t.x; p[1] = (_Float16)t.y;
    p[2] = (_Float16)t.z; p[3] = (_Float16)t.w;
  }
}

// ---------------------------------------------------------------------------
// Kernel 1: projection GEMM  out[m][n] = X[m][:] . W[n][:] + bias[n]
// X: [4096,1024] f16, W: [1024,1024] f16 (pre-converted), f16 output.
// VLAYOUT==0 -> [B,H,S,D] (Q,K);  VLAYOUT==1 -> [B,H,D,S] (V, pre-transposed).
// Block: 256 threads (8 waves), tile 256x64, wave tile 32x64, K-step 32.
// Double-buffered LDS, all staging via async copies overlapped with WMMAs.
// ---------------------------------------------------------------------------
template <int VLAYOUT>
__global__ __launch_bounds__(256) void proj_gemm_kernel(
    const _Float16* __restrict__ X, const _Float16* __restrict__ W,
    const float* __restrict__ bias, _Float16* __restrict__ out) {
  __shared__ _Float16 As[2][256 * 32];  // [m][k] 2 x 16 KB
  __shared__ _Float16 Bs[2][64 * 32];   // [n][k] 2 x  4 KB
  const int tid = threadIdx.x;
  const int wave = tid >> 5;
  const int lane = tid & 31;
  const int hs = lane >> 4;
  const int row = lane & 15;
  const int m_blk = blockIdx.x * 256;
  const int n_blk = blockIdx.y * 64;

  auto stage = [&](int buf, int kk) {
#pragma unroll
    for (int r = 0; r < 4; ++r) {  // A: 1024 x 16B chunks / 256 thr
      int i = tid + r * 256;
      int m = i >> 2, k8 = (i & 3) << 3;
      async_copy16(&As[buf][m * 32 + k8],
                   X + (size_t)(m_blk + m) * CQ + kk + k8);
    }
    {  // B: 256 x 16B chunks / 256 thr
      int n = tid >> 2, k8 = (tid & 3) << 3;
      async_copy16(&Bs[buf][n * 32 + k8],
                   W + (size_t)(n_blk + n) * CQ + kk + k8);
    }
  };

  stage(0, 0);
  wait_async0();
  __syncthreads();

  v8f acc[2][4] = {};
  int buf = 0;
  for (int kk = 0; kk < CQ; kk += 32) {
    if (kk + 32 < CQ) stage(buf ^ 1, kk + 32);  // overlap with WMMAs below
    const int mb = wave * 32;
    v16h af[2];
#pragma unroll
    for (int mi = 0; mi < 2; ++mi) {
      v8h a0 = *(const v8h*)&As[buf][(mb + mi * 16 + row) * 32 + hs * 16];
      v8h a1 = *(const v8h*)&As[buf][(mb + mi * 16 + row) * 32 + hs * 16 + 8];
      af[mi] = make_frag(a0, a1);
    }
    v16h bf[4];
#pragma unroll
    for (int nt = 0; nt < 4; ++nt) {
      v8h b0 = *(const v8h*)&Bs[buf][(nt * 16 + row) * 32 + hs * 16];
      v8h b1 = *(const v8h*)&Bs[buf][(nt * 16 + row) * 32 + hs * 16 + 8];
      bf[nt] = make_frag(b0, b1);
    }
#pragma unroll
    for (int mi = 0; mi < 2; ++mi)
#pragma unroll
      for (int nt = 0; nt < 4; ++nt)
        acc[mi][nt] = wmma_f16(af[mi], bf[nt], acc[mi][nt]);
    wait_async0();
    __syncthreads();
    buf ^= 1;
  }
  // ---- epilogue: +bias, f16 store head-major (optionally pre-transposed) ----
#pragma unroll
  for (int mi = 0; mi < 2; ++mi)
#pragma unroll
    for (int nt = 0; nt < 4; ++nt)
#pragma unroll
      for (int v = 0; v < 8; ++v) {
        int m = m_blk + wave * 32 + mi * 16 + v + hs * 8;
        int n = n_blk + nt * 16 + row;
        float val = acc[mi][nt][v] + bias[n];
        int b = m >> 10, s = m & 1023, h = n >> 6, d = n & 63;
        if (VLAYOUT)
          out[(((size_t)(b * HQ + h)) * DQ + d) * 1024 + s] = (_Float16)val;
        else
          out[(((size_t)(b * HQ + h)) * 1024 + s) * DQ + d] = (_Float16)val;
      }
}

// ---------------------------------------------------------------------------
// Kernel 2: flash attention per (b,h). Block: 128 threads (4 waves x 16 rows).
// Kh: [B,H,T,D] f16, Vt: [B,H,D,T] f16 (pre-transposed by projection).
// Double-buffered K/V tiles staged by the async engine during compute.
// ---------------------------------------------------------------------------
__global__ __launch_bounds__(128) void attn_kernel(
    const _Float16* __restrict__ Qh, const _Float16* __restrict__ Kh,
    const _Float16* __restrict__ Vt, const float* __restrict__ rel_bias,
    const unsigned char* __restrict__ mask, _Float16* __restrict__ O) {
  __shared__ _Float16 Ks[2][32 * 64];   // [t][d]
  __shared__ _Float16 Vts[2][64 * 32];  // [d][t]
  __shared__ _Float16 Ps[4 * 16 * 32];  // per-wave P staging [f][t]

  const int tid = threadIdx.x;
  const int wave = tid >> 5;
  const int lane = tid & 31;
  const int hs = lane >> 4;
  const int row = lane & 15;
  const int bh = blockIdx.y;
  const int b = bh >> 4;
  const int h = bh & 15;
  const int f0 = blockIdx.x * 64 + wave * 16;

  const _Float16* kbh = Kh + (size_t)bh * TQ * DQ;
  const _Float16* vbh = Vt + (size_t)bh * DQ * TQ;

  auto stageKV = [&](int buf, int t0) {
#pragma unroll
    for (int i = tid; i < 256; i += 128) {  // K tile [t][d]: 32x64
      int t = i >> 3, d8 = (i & 7) << 3;
      async_copy16(&Ks[buf][t * 64 + d8], kbh + (size_t)(t0 + t) * DQ + d8);
    }
#pragma unroll
    for (int i = tid; i < 256; i += 128) {  // V tile [d][t]: 64x32
      int d = i >> 2, t8 = (i & 3) << 3;
      async_copy16(&Vts[buf][d * 32 + t8], vbh + (size_t)d * TQ + t0 + t8);
    }
  };

  // Q: 16x64 f16 = two 16x32 A-frags, loaded once
  const _Float16* qbase = Qh + ((size_t)bh * FQ + (f0 + row)) * DQ;
  v16h qa[2];
#pragma unroll
  for (int c = 0; c < 2; ++c) {
    v8h q0 = *(const v8h*)(qbase + c * 32 + hs * 16);
    v8h q1 = *(const v8h*)(qbase + c * 32 + hs * 16 + 8);
    qa[c] = make_frag(q0, q1);
  }

  float m_run[8], l_run[8];
#pragma unroll
  for (int v = 0; v < 8; ++v) { m_run[v] = -1e30f; l_run[v] = 0.f; }
  v8f oacc[4] = {};
  _Float16* myP = &Ps[wave * 512];

  stageKV(0, 0);
  wait_async0();
  __syncthreads();

  int buf = 0;
  for (int t0 = 0; t0 < TQ; t0 += 32) {
    if (t0 + 32 < TQ) stageKV(buf ^ 1, t0 + 32);  // overlap with compute

    // S = Q @ K^T : two 16x16 t-tiles, contraction over D=64 (2 WMMAs each)
    v8f s[2];
#pragma unroll
    for (int j = 0; j < 2; ++j) {
      v8f a = {};
#pragma unroll
      for (int c = 0; c < 2; ++c) {
        v8h b0 =
            *(const v8h*)&Ks[buf][(j * 16 + row) * 64 + c * 32 + hs * 16];
        v8h b1 =
            *(const v8h*)&Ks[buf][(j * 16 + row) * 64 + c * 32 + hs * 16 + 8];
        a = wmma_f16(qa[c], make_frag(b0, b1), a);
      }
      s[j] = a;
    }

    // scale 1/sqrt(64), + rel_bias, mask==True -> -10000 (reference order)
#pragma unroll
    for (int j = 0; j < 2; ++j) {
      const int tcol = t0 + j * 16 + row;
#pragma unroll
      for (int v = 0; v < 8; ++v) {
        const int frow = f0 + v + hs * 8;
        float val = s[j][v] * 0.125f +
                    rel_bias[((size_t)h * FQ + frow) * TQ + tcol];
        if (mask[((size_t)b * FQ + frow) * TQ + tcol]) val = -10000.f;
        s[j][v] = val;
      }
    }

    // online softmax: rows live across 16-lane halves -> xor-shuffle reduce
    float alpha[8];
#pragma unroll
    for (int v = 0; v < 8; ++v) {
      float rmax = fmaxf(s[0][v], s[1][v]);
#pragma unroll
      for (int off = 8; off > 0; off >>= 1)
        rmax = fmaxf(rmax, __shfl_xor(rmax, off));
      float mn = fmaxf(m_run[v], rmax);
      alpha[v] = __expf(m_run[v] - mn);
      float p0 = __expf(s[0][v] - mn);
      float p1 = __expf(s[1][v] - mn);
      s[0][v] = p0; s[1][v] = p1;
      float rs = p0 + p1;
#pragma unroll
      for (int off = 8; off > 0; off >>= 1) rs += __shfl_xor(rs, off);
      l_run[v] = l_run[v] * alpha[v] + rs;
      m_run[v] = mn;
    }
#pragma unroll
    for (int nt = 0; nt < 4; ++nt)
#pragma unroll
      for (int v = 0; v < 8; ++v) oacc[nt][v] *= alpha[v];

    // P: C-frag layout -> LDS [f][t] -> reload as A-frag (cross-lane dep)
#pragma unroll
    for (int j = 0; j < 2; ++j)
#pragma unroll
      for (int v = 0; v < 8; ++v)
        myP[(v + hs * 8) * 32 + j * 16 + row] = (_Float16)s[j][v];
    __builtin_amdgcn_wave_barrier();
    asm volatile("s_wait_dscnt 0" ::: "memory");
    v8h p0 = *(const v8h*)&myP[row * 32 + hs * 16];
    v8h p1 = *(const v8h*)&myP[row * 32 + hs * 16 + 8];
    v16h pa = make_frag(p0, p1);

    // O += P @ V : four 16x16 d-tiles, contraction over t-chunk = 32
#pragma unroll
    for (int nt = 0; nt < 4; ++nt) {
      v8h b0 = *(const v8h*)&Vts[buf][(nt * 16 + row) * 32 + hs * 16];
      v8h b1 = *(const v8h*)&Vts[buf][(nt * 16 + row) * 32 + hs * 16 + 8];
      oacc[nt] = wmma_f16(pa, make_frag(b0, b1), oacc[nt]);
    }

    wait_async0();
    __syncthreads();
    buf ^= 1;
  }

  // normalize and store f16 O in [B,F,C] (C = h*64 + d)
#pragma unroll
  for (int v = 0; v < 8; ++v) l_run[v] = 1.0f / l_run[v];
#pragma unroll
  for (int nt = 0; nt < 4; ++nt)
#pragma unroll
    for (int v = 0; v < 8; ++v) {
      int frow = f0 + v + hs * 8;
      O[((size_t)b * FQ + frow) * CQ + h * DQ + nt * 16 + row] =
          (_Float16)(oacc[nt][v] * l_run[v]);
    }
}

// ---------------------------------------------------------------------------
// Kernel 3: output projection  d_out[m][n] = Oh[m][:] . Wo[n][:] + bo[n]
// Both operands f16 -> fully async staging, double-buffered, tile 256x64.
// ---------------------------------------------------------------------------
__global__ __launch_bounds__(256) void out_gemm_kernel(
    const _Float16* __restrict__ A, const _Float16* __restrict__ W,
    const float* __restrict__ bias, float* __restrict__ out) {
  __shared__ _Float16 As[2][256 * 32];
  __shared__ _Float16 Bs[2][64 * 32];
  const int tid = threadIdx.x;
  const int wave = tid >> 5;
  const int lane = tid & 31;
  const int hs = lane >> 4;
  const int row = lane & 15;
  const int m_blk = blockIdx.x * 256;
  const int n_blk = blockIdx.y * 64;

  auto stage = [&](int buf, int kk) {
#pragma unroll
    for (int r = 0; r < 4; ++r) {
      int i = tid + r * 256;
      int m = i >> 2, k8 = (i & 3) << 3;
      async_copy16(&As[buf][m * 32 + k8],
                   A + (size_t)(m_blk + m) * CQ + kk + k8);
    }
    {
      int n = tid >> 2, k8 = (tid & 3) << 3;
      async_copy16(&Bs[buf][n * 32 + k8],
                   W + (size_t)(n_blk + n) * CQ + kk + k8);
    }
  };

  stage(0, 0);
  wait_async0();
  __syncthreads();

  v8f acc[2][4] = {};
  int buf = 0;
  for (int kk = 0; kk < CQ; kk += 32) {
    if (kk + 32 < CQ) stage(buf ^ 1, kk + 32);
    const int mb = wave * 32;
    v16h af[2];
#pragma unroll
    for (int mi = 0; mi < 2; ++mi) {
      v8h a0 = *(const v8h*)&As[buf][(mb + mi * 16 + row) * 32 + hs * 16];
      v8h a1 = *(const v8h*)&As[buf][(mb + mi * 16 + row) * 32 + hs * 16 + 8];
      af[mi] = make_frag(a0, a1);
    }
    v16h bf[4];
#pragma unroll
    for (int nt = 0; nt < 4; ++nt) {
      v8h b0 = *(const v8h*)&Bs[buf][(nt * 16 + row) * 32 + hs * 16];
      v8h b1 = *(const v8h*)&Bs[buf][(nt * 16 + row) * 32 + hs * 16 + 8];
      bf[nt] = make_frag(b0, b1);
    }
#pragma unroll
    for (int mi = 0; mi < 2; ++mi)
#pragma unroll
      for (int nt = 0; nt < 4; ++nt)
        acc[mi][nt] = wmma_f16(af[mi], bf[nt], acc[mi][nt]);
    wait_async0();
    __syncthreads();
    buf ^= 1;
  }
#pragma unroll
  for (int mi = 0; mi < 2; ++mi)
#pragma unroll
    for (int nt = 0; nt < 4; ++nt)
#pragma unroll
      for (int v = 0; v < 8; ++v) {
        int m = m_blk + wave * 32 + mi * 16 + v + hs * 8;
        int n = n_blk + nt * 16 + row;
        out[(size_t)m * CQ + n] = acc[mi][nt][v] + bias[n];
      }
}

// ---------------------------------------------------------------------------
extern "C" void kernel_launch(void* const* d_in, const int* in_sizes, int n_in,
                              void* d_out, int out_size, void* d_ws,
                              size_t ws_size, hipStream_t stream) {
  (void)in_sizes; (void)n_in; (void)out_size; (void)ws_size;
  const float* q  = (const float*)d_in[0];
  const float* k  = (const float*)d_in[1];
  const float* v  = (const float*)d_in[2];
  const unsigned char* mask = (const unsigned char*)d_in[3];  // bool tensor
  const float* rb = (const float*)d_in[4];
  const float* Wq = (const float*)d_in[5];
  const float* bq = (const float*)d_in[6];
  const float* Wk = (const float*)d_in[7];
  const float* bk = (const float*)d_in[8];
  const float* Wv = (const float*)d_in[9];
  const float* bv = (const float*)d_in[10];
  const float* Wo = (const float*)d_in[11];
  const float* bo = (const float*)d_in[12];
  float* out = (float*)d_out;

  const size_t NIN = (size_t)4096 * 1024;  // input matrix elements
  const size_t NW  = (size_t)1024 * 1024;  // weight matrix elements
  _Float16* Xq  = (_Float16*)d_ws;         // f16 q      8 MB (reused as Oh)
  _Float16* Xk  = Xq + NIN;                // f16 k      8 MB
  _Float16* Xv  = Xk + NIN;                // f16 v      8 MB
  _Float16* Whq = Xv + NIN;                // f16 Wq     2 MB
  _Float16* Whk = Whq + NW;                // f16 Wk     2 MB
  _Float16* Whv = Whk + NW;                // f16 Wv     2 MB
  _Float16* Who = Whv + NW;                // f16 Wo     2 MB
  _Float16* Qh  = Who + NW;                // [B,H,F,D]  8 MB
  _Float16* Kh  = Qh + NIN;                // [B,H,T,D]  8 MB
  _Float16* Vt  = Kh + NIN;                // [B,H,D,T]  8 MB
  _Float16* Oh  = Xq;  // O aliases dead Xq (safe: stream-ordered)

  // one-shot f32 -> f16 conversions
  cvt_f16_kernel<<<2048, 256, 0, stream>>>(q, Xq, (int)(NIN / 4));
  cvt_f16_kernel<<<2048, 256, 0, stream>>>(k, Xk, (int)(NIN / 4));
  cvt_f16_kernel<<<2048, 256, 0, stream>>>(v, Xv, (int)(NIN / 4));
  cvt_f16_kernel<<<1024, 256, 0, stream>>>(Wq, Whq, (int)(NW / 4));
  cvt_f16_kernel<<<1024, 256, 0, stream>>>(Wk, Whk, (int)(NW / 4));
  cvt_f16_kernel<<<1024, 256, 0, stream>>>(Wv, Whv, (int)(NW / 4));
  cvt_f16_kernel<<<1024, 256, 0, stream>>>(Wo, Who, (int)(NW / 4));

  dim3 gproj(16, 16, 1);  // 4096/256 x 1024/64
  proj_gemm_kernel<0><<<gproj, 256, 0, stream>>>(Xq, Whq, bq, Qh);
  proj_gemm_kernel<0><<<gproj, 256, 0, stream>>>(Xk, Whk, bk, Kh);
  proj_gemm_kernel<1><<<gproj, 256, 0, stream>>>(Xv, Whv, bv, Vt);
  attn_kernel<<<dim3(16, 64, 1), 128, 0, stream>>>(Qh, Kh, Vt, rb, mask, Oh);
  out_gemm_kernel<<<gproj, 256, 0, stream>>>(Oh, Who, bo, out);
}
